// GNN_EP_40613210751436
// MI455X (gfx1250) — compile-verified, matrix operand order checked by reading, not run on hardware
//
#include <hip/hip_runtime.h>
#include <hip/hip_bf16.h>
#include <math.h>

// ---------------------------------------------------------------------------
// Types for CDNA5 WMMA
// ---------------------------------------------------------------------------
typedef __bf16 bf16;
typedef __attribute__((ext_vector_type(16))) __bf16 v16bf;
typedef __attribute__((ext_vector_type(8)))  __bf16 v8bf;
typedef __attribute__((ext_vector_type(4)))  __bf16 v4bf;
typedef __attribute__((ext_vector_type(8)))  float  v8f;

#define DMODEL 768
#define NB     8
#define NN     64
#define NHEADS 12
#define DH     64
#define NLAY   6
#define CATW   128
#define FROWS  (NB*NN)        // 512
#define EROWS  (NB*NN*NN)     // 32768

// GEMM tiling
#define BM 64
#define BN 64
#define BK 32
#define LDS_STRIDE (BK + 8)   // 40 bf16 -> 80B rows, 8/16B aligned subruns

#define FLAG_SILU  1
#define FLAG_ACCUM 2

__device__ __forceinline__ float silu_f(float x) {
    return x * (1.0f / (1.0f + __expf(-x)));
}

// ---------------------------------------------------------------------------
// Weight convert + transpose: Wt[n*K + k] = (bf16) W[k*N + n]
// ---------------------------------------------------------------------------
__global__ __launch_bounds__(256)
void k_wcvt(const float* __restrict__ W, bf16* __restrict__ Wt, int K, int N)
{
    __shared__ float tile[32][33];
    const int kb = blockIdx.x * 32, nb = blockIdx.y * 32;
    const int tx = threadIdx.x & 31, ty = threadIdx.x >> 5;   // 32 x 8
    for (int r = ty; r < 32; r += 8) {
        int gk = kb + r, gn = nb + tx;
        tile[r][tx] = (gk < K && gn < N) ? W[(size_t)gk * N + gn] : 0.0f;
    }
    __syncthreads();
    for (int r = ty; r < 32; r += 8) {
        int gn = nb + r, gk = kb + tx;
        if (gn < N && gk < K) Wt[(size_t)gn * K + gk] = (bf16)tile[tx][r];
    }
}

// ---------------------------------------------------------------------------
// Generic WMMA GEMM:  C[M,N] (ldC) = act( A[M,K] @ W[K,N] + bias )  [+= C]
// A fp32 (staged to LDS as bf16 via b128 loads + cvt_pk_bf16);
// W pre-converted bf16, transposed [N][K] -> B tile staged with
// global_load_async_to_lds_b64 (ASYNCcnt) when fully in-bounds.
// Block: 128 threads = 4 waves; 64x64 tile; wave w owns 16 N-columns.
// ---------------------------------------------------------------------------
__global__ __launch_bounds__(128)
void k_gemm(float* __restrict__ C, int ldC,
            const float* __restrict__ A,
            const bf16* __restrict__ Wt,     // [N][K]
            const float* __restrict__ bias,
            int M, int K, int N, int flags)
{
    __shared__ __align__(16) bf16 As[BM][LDS_STRIDE];   // As[m][k]
    __shared__ __align__(16) bf16 Bs[BN][LDS_STRIDE];   // Bs[n][k]

    const int tid  = threadIdx.x;
    const int wave = tid >> 5;
    const int lane = tid & 31;
    const int lh   = lane & 15;
    const int kh   = lane >> 4;
    const int m0   = blockIdx.x * BM;
    const int n0   = blockIdx.y * BN;
    const bool mFull = (m0 + BM <= M);
    const bool nFull = (n0 + BN <= N);
    const unsigned bsBase = (unsigned)(uintptr_t)(&Bs[0][0]);

    v8f acc[4];
#pragma unroll
    for (int i = 0; i < 4; ++i)
#pragma unroll
        for (int j = 0; j < 8; ++j) acc[i][j] = 0.0f;

    for (int k0 = 0; k0 < K; k0 += BK) {
        const bool kFull = (k0 + BK <= K);
        bool usedAsync = false;

        // ---- stage W tile: Bs[n][k] = Wt[n0+n][k0+k] ----
        if (nFull && kFull) {
#if defined(__gfx1250__)
            usedAsync = true;
#pragma unroll
            for (int i = 0; i < 4; ++i) {
                int vi = tid + i * 128;               // 0..511
                int n = vi >> 3, part = vi & 7;       // 8 bytes each
                unsigned laddr = bsBase + (unsigned)(n * (LDS_STRIDE * 2) + part * 8);
                const bf16* gp = Wt + (size_t)(n0 + n) * K + k0 + part * 4;
                asm volatile("global_load_async_to_lds_b64 %0, %1, off"
                             :: "v"(laddr), "v"((unsigned long long)(uintptr_t)gp)
                             : "memory");
            }
#else
#pragma unroll
            for (int i = 0; i < 4; ++i) {
                int vi = tid + i * 128;
                int n = vi >> 3, part = vi & 7;
                const bf16* gp = Wt + (size_t)(n0 + n) * K + k0 + part * 4;
                *(v4bf*)&Bs[n][part * 4] = *(const v4bf*)gp;
            }
#endif
        } else {
            for (int x = tid; x < BK * BN; x += 128) {
                int r = x & 31, c = x >> 5;           // r=k (coalesced), c=n
                int gk = k0 + r, gn = n0 + c;
                bf16 v = (bf16)0.0f;
                if (gk < K && gn < N) v = Wt[(size_t)gn * K + gk];
                Bs[c][r] = v;
            }
        }

        // ---- stage A tile: As[m][k] = bf16(A[m0+m][k0+k]) ----
        if (mFull && kFull) {
#pragma unroll
            for (int i = 0; i < 4; ++i) {
                int vi = tid + i * 128;               // 0..511
                int r = vi >> 3, c4 = vi & 7;         // float4 each
                const float4 f = *(const float4*)(A + (size_t)(m0 + r) * K + k0 + c4 * 4);
                v4bf b;
                b[0] = (bf16)f.x; b[1] = (bf16)f.y; b[2] = (bf16)f.z; b[3] = (bf16)f.w;
                *(v4bf*)&As[r][c4 * 4] = b;
            }
        } else {
            for (int x = tid; x < BM * BK; x += 128) {
                int r = x >> 5, c = x & 31;
                int gm = m0 + r, gk = k0 + c;
                float v = (gm < M && gk < K) ? A[(size_t)gm * K + gk] : 0.0f;
                As[r][c] = (bf16)v;
            }
        }

#if defined(__gfx1250__)
        if (usedAsync) asm volatile("s_wait_asynccnt 0x0" ::: "memory");
#endif
        __syncthreads();

        if (k0 + BK < K) {   // prefetch next A tile (global_prefetch_b8)
            __builtin_prefetch(&A[(size_t)(m0 + (tid >> 1)) * K + k0 + BK], 0, 0);
        }

        // ---- B fragment for this wave's 16-column stripe ----
        v16bf bfrag;
        {
            const bf16* bp = &Bs[wave * 16 + lh][0];
            v8bf lo = *(const v8bf*)(bp + kh * 8);
            v8bf hi = *(const v8bf*)(bp + 16 + kh * 8);
#pragma unroll
            for (int i = 0; i < 8; ++i) { bfrag[i] = lo[i]; bfrag[i + 8] = hi[i]; }
        }
        // ---- 4 M-subtiles ----
#pragma unroll
        for (int mi = 0; mi < 4; ++mi) {
            const bf16* ap = &As[mi * 16 + lh][0];
            v8bf lo = *(const v8bf*)(ap + kh * 8);
            v8bf hi = *(const v8bf*)(ap + 16 + kh * 8);
            v16bf afrag;
#pragma unroll
            for (int i = 0; i < 8; ++i) { afrag[i] = lo[i]; afrag[i + 8] = hi[i]; }
#if defined(__gfx1250__)
            acc[mi] = __builtin_amdgcn_wmma_f32_16x16x32_bf16(
                false, afrag, false, bfrag, (short)0, acc[mi], false, false);
#else
            for (int i = 0; i < 8; ++i) acc[mi][i] += (float)afrag[i] * (float)bfrag[i];
#endif
        }
        __syncthreads();
    }

    // ---- epilogue ----
    const int n = n0 + wave * 16 + lh;
    const float bv = (bias != nullptr && n < N) ? bias[n] : 0.0f;
#pragma unroll
    for (int mi = 0; mi < 4; ++mi) {
#pragma unroll
        for (int j = 0; j < 8; ++j) {
            int m = m0 + mi * 16 + kh * 8 + j;
            if (m < M && n < N) {
                float v = acc[mi][j] + bv;
                if (flags & FLAG_SILU) v = silu_f(v);
                size_t idx = (size_t)m * ldC + n;
                if (flags & FLAG_ACCUM) v += C[idx];
                C[idx] = v;
            }
        }
    }
}

// ---------------------------------------------------------------------------
// LayerNorm (one row per block): out = LN(x [+res]) * g + b, optional SiLU,
// optional per-row mask multiply. d <= 1536.
// ---------------------------------------------------------------------------
__global__ __launch_bounds__(256)
void k_layernorm(float* __restrict__ out, const float* __restrict__ x,
                 const float* __restrict__ res,
                 const float* __restrict__ g, const float* __restrict__ b,
                 const float* __restrict__ rowmask, int d, int do_silu)
{
    __shared__ float buf[1536];
    __shared__ float red[256];
    const int row = blockIdx.x;
    const int tid = threadIdx.x;
    const size_t base = (size_t)row * d;

    float s1 = 0.0f, s2 = 0.0f;
    for (int i = tid; i < d; i += 256) {
        float v = x[base + i] + (res ? res[base + i] : 0.0f);
        buf[i] = v; s1 += v; s2 += v * v;
    }
    red[tid] = s1; __syncthreads();
    for (int o = 128; o > 0; o >>= 1) { if (tid < o) red[tid] += red[tid + o]; __syncthreads(); }
    float mean = red[0] / (float)d;
    __syncthreads();
    red[tid] = s2; __syncthreads();
    for (int o = 128; o > 0; o >>= 1) { if (tid < o) red[tid] += red[tid + o]; __syncthreads(); }
    float var = red[0] / (float)d - mean * mean;
    float inv = rsqrtf(var + 1e-5f);
    float mk = rowmask ? rowmask[row] : 1.0f;

    for (int i = tid; i < d; i += 256) {
        float v = (buf[i] - mean) * inv * g[i] + b[i];
        if (do_silu) v = silu_f(v);
        out[base + i] = v * mk;
    }
}

// hin[b,i,:] = h[b,i,:] + sum_j relu(h[b,j,:] + e[b,i,j,:]) * emask[b,i,j]
__global__ __launch_bounds__(256)
void k_agg(float* __restrict__ hin, const float* __restrict__ h,
           const float* __restrict__ e, const float* __restrict__ emask)
{
    const int fi = blockIdx.x;        // b*N + i
    const int b  = fi >> 6;
    for (int d = threadIdx.x; d < DMODEL; d += 256) {
        float acc = h[(size_t)fi * DMODEL + d];
        for (int j = 0; j < NN; ++j) {
            float m = emask[((size_t)fi << 6) + j];
            float v = h[(size_t)((b << 6) + j) * DMODEL + d]
                    + e[(((size_t)fi << 6) + j) * DMODEL + d];
            acc += fmaxf(v, 0.0f) * m;
        }
        hin[(size_t)fi * DMODEL + d] = acc;
    }
}

// dense_ez[b,i,j,:] = ((surf_i + surf_j)/2 + edge[b,i,j,:] + temb[b,:]) * emask
__global__ __launch_bounds__(256)
void k_build_dense_ez(float* __restrict__ dz, const float* __restrict__ surf,
                      const float* __restrict__ edg, const float* __restrict__ temb,
                      const float* __restrict__ emask)
{
    const int row = blockIdx.x;
    const int b = row >> 12;
    const int i = (row >> 6) & 63;
    const int j = row & 63;
    const float m = emask[row];
    const float* si = surf + (size_t)((b << 6) + i) * DMODEL;
    const float* sj = surf + (size_t)((b << 6) + j) * DMODEL;
    const float* tb = temb + (size_t)b * DMODEL;
    for (int d = threadIdx.x; d < DMODEL; d += 256) {
        float v = (si[d] + sj[d]) * 0.5f + edg[(size_t)row * DMODEL + d] + tb[d];
        dz[(size_t)row * DMODEL + d] = v * m;
    }
}

__global__ void k_add2(float* out, const float* a, const float* b, size_t n)
{
    size_t i = (size_t)blockIdx.x * blockDim.x + threadIdx.x;
    size_t stride = (size_t)gridDim.x * blockDim.x;
    for (; i < n; i += stride) out[i] = a[i] + b[i];
}

// h[b,i,:] = surf[b,i,:] + temb[b,:]
__global__ __launch_bounds__(256)
void k_add_bcastB(float* __restrict__ out, const float* __restrict__ a,
                  const float* __restrict__ temb)
{
    const int fi = blockIdx.x;
    const int b = fi >> 6;
    for (int d = threadIdx.x; d < DMODEL; d += 256)
        out[(size_t)fi * DMODEL + d] = a[(size_t)fi * DMODEL + d] + temb[(size_t)b * DMODEL + d];
}

// Attention: one block per (b, head, query). 64 threads.
__global__ __launch_bounds__(64)
void k_attention(const float* __restrict__ q, const float* __restrict__ k,
                 const float* __restrict__ v, const float* __restrict__ fm,
                 float* __restrict__ o)
{
    const int idx = blockIdx.x;            // ((b*NH)+h)*N + qi
    const int qi = idx & 63;
    const int bh = idx >> 6;
    const int hh = bh % NHEADS;
    const int b  = bh / NHEADS;
    const int t  = threadIdx.x;

    __shared__ float sc[NN];
    __shared__ float red[NN];

    const float* qp = q + (size_t)((b << 6) + qi) * DMODEL + hh * DH;
    const float* kp = k + (size_t)((b << 6) + t ) * DMODEL + hh * DH;
    float s = 0.0f;
    for (int d = 0; d < DH; ++d) s += qp[d] * kp[d];
    s *= 0.125f;
    if (!(fm[(b << 6) + t] > 0.0f)) s = -1e9f;
    sc[t] = s; red[t] = s; __syncthreads();
    for (int off = 32; off > 0; off >>= 1) { if (t < off) red[t] = fmaxf(red[t], red[t + off]); __syncthreads(); }
    float mx = red[0]; __syncthreads();
    float ex = __expf(sc[t] - mx);
    sc[t] = ex; red[t] = ex; __syncthreads();
    for (int off = 32; off > 0; off >>= 1) { if (t < off) red[t] += red[t + off]; __syncthreads(); }
    float inv = 1.0f / red[0];
    __syncthreads();

    float acc = 0.0f;
    for (int kk = 0; kk < NN; ++kk)
        acc += sc[kk] * v[(size_t)((b << 6) + kk) * DMODEL + hh * DH + t];
    o[(size_t)((b << 6) + qi) * DMODEL + hh * DH + t] = acc * inv;
}

// e = LN(lne; e + silu(we1(e) + u_i + u_j + bwe2)) * emask   (one block/row)
__global__ __launch_bounds__(256)
void k_edge_update(float* __restrict__ e, const float* __restrict__ t0,
                   const float* __restrict__ u, const float* __restrict__ bwe2,
                   const float* __restrict__ emask,
                   const float* __restrict__ g, const float* __restrict__ bb)
{
    __shared__ float buf[DMODEL];
    __shared__ float red[256];
    const int row = blockIdx.x;
    const int b = row >> 12;
    const int i = (row >> 6) & 63;
    const int j = row & 63;
    const int tid = threadIdx.x;
    const float* ui = u + (size_t)((b << 6) + i) * DMODEL;
    const float* uj = u + (size_t)((b << 6) + j) * DMODEL;
    const size_t base = (size_t)row * DMODEL;

    float s1 = 0.0f, s2 = 0.0f;
    for (int d = tid; d < DMODEL; d += 256) {
        float pre = t0[base + d] + ui[d] + uj[d] + bwe2[d];
        float xv = e[base + d] + silu_f(pre);
        buf[d] = xv; s1 += xv; s2 += xv * xv;
    }
    red[tid] = s1; __syncthreads();
    for (int o = 128; o > 0; o >>= 1) { if (tid < o) red[tid] += red[tid + o]; __syncthreads(); }
    float mean = red[0] / (float)DMODEL;
    __syncthreads();
    red[tid] = s2; __syncthreads();
    for (int o = 128; o > 0; o >>= 1) { if (tid < o) red[tid] += red[tid + o]; __syncthreads(); }
    float var = red[0] / (float)DMODEL - mean * mean;
    float inv = rsqrtf(var + 1e-5f);
    float mk = emask[row];
    for (int d = tid; d < DMODEL; d += 256)
        e[base + d] = ((buf[d] - mean) * inv * g[d] + bb[d]) * mk;
}

// t1[row,:] = mask*(t0[row,:] + vloc_i + vloc_j) + b1
__global__ __launch_bounds__(256)
void k_fc2_combine(float* __restrict__ t1, const float* __restrict__ t0,
                   const float* __restrict__ vloc, const float* __restrict__ b1,
                   const float* __restrict__ emask)
{
    const int row = blockIdx.x;
    const int b = row >> 12;
    const int i = (row >> 6) & 63;
    const int j = row & 63;
    const float mk = emask[row];
    const float* vi = vloc + (size_t)((b << 6) + i) * DMODEL;
    const float* vj = vloc + (size_t)((b << 6) + j) * DMODEL;
    const size_t base = (size_t)row * DMODEL;
    for (int d = threadIdx.x; d < DMODEL; d += 256)
        t1[base + d] = mk * (t0[base + d] + vi[d] + vj[d]) + b1[d];
}

// sincos timestep embedding
__global__ __launch_bounds__(384)
void k_sincos(const int* __restrict__ ts, float* __restrict__ sc)
{
    const int b = blockIdx.x;
    const int i = threadIdx.x;                 // 0..383
    float f = __expf(-logf(10000.0f) * (float)i / 384.0f);
    float a = (float)ts[b] * f;
    sc[(size_t)b * DMODEL + i]       = sinf(a);
    sc[(size_t)b * DMODEL + 384 + i] = cosf(a);
}

// out[b,i,j,c] = 0.5*(pred[b,i,j,c]*m[b,i,j] + pred[b,j,i,c]*m[b,j,i])
__global__ void k_final(float* __restrict__ out, const float* __restrict__ pred,
                        const float* __restrict__ emask)
{
    size_t idx = (size_t)blockIdx.x * blockDim.x + threadIdx.x;
    size_t total = (size_t)EROWS * 6;
    size_t stride = (size_t)gridDim.x * blockDim.x;
    for (; idx < total; idx += stride) {
        int row = (int)(idx / 6), c = (int)(idx % 6);
        int b = row >> 12, i = (row >> 6) & 63, j = row & 63;
        int rowT = (b << 12) | (j << 6) | i;
        out[idx] = 0.5f * (pred[(size_t)row * 6 + c] * emask[row] +
                           pred[(size_t)rowT * 6 + c] * emask[rowT]);
    }
}

// ---------------------------------------------------------------------------
// Host side. jax pytree (sorted-key) flatten order:
// 0 edgePos, 1 edge_mask, 2 face_mask, [3..236 params], 237 surfPos,
// 238 surfZ, 239 timesteps.
// seq4 leaves: l1.b, l1.w, l2.b, l2.w, ln.b, ln.g
// params: edgep_embed(3), fc_out(9), fc_out2(15), fc_pool(21),
//   layers(27+32*l), surfp_embed(219), surfz_embed(225), time_embed(231)
// ---------------------------------------------------------------------------
extern "C" void kernel_launch(void* const* d_in, const int* in_sizes, int n_in,
                              void* d_out, int out_size, void* d_ws, size_t ws_size,
                              hipStream_t stream)
{
    (void)in_sizes; (void)n_in; (void)out_size; (void)ws_size;
    auto P = [&](int i) -> const float* { return (const float*)d_in[i]; };

    const float* edgePos  = P(0);
    const float* emask    = P(1);
    const float* fmask    = P(2);
    const float* surfPos  = P(237);
    const float* surfZ    = P(238);
    const int*   tsteps   = (const int*)d_in[239];
    float* out = (float*)d_out;

    const int IDX_EDGEP = 3, IDX_FCOUT = 9, IDX_FCOUT2 = 15, IDX_FCPOOL = 21;
    const int IDX_LAYERS = 27, IDX_SURFP = 219, IDX_SURFZ = 225, IDX_TIME = 231;

    char* wsb = (char*)d_ws;
    size_t off = 0;
    auto allocf = [&](size_t elems) -> float* {
        float* p = (float*)(wsb + off);
        off += elems * sizeof(float);
        off = (off + 255) & ~(size_t)255;
        return p;
    };
    auto allocb = [&](size_t elems) -> bf16* {
        bf16* p = (bf16*)(wsb + off);
        off += elems * sizeof(bf16);
        off = (off + 255) & ~(size_t)255;
        return p;
    };
    const size_t ED = (size_t)EROWS * DMODEL;
    const size_t FD = (size_t)FROWS * DMODEL;

    float* e        = allocf(ED);
    float* dense_ez = allocf(ED);
    float* eh       = allocf(ED);
    float* t0       = allocf(ED);
    float* t1       = allocf(ED);
    float* surf     = allocf(FD);
    float* h        = allocf(FD);
    float* hin      = allocf(FD);
    float* qb       = allocf(FD);
    float* kb       = allocf(FD);
    float* vb       = allocf(FD);
    float* ob       = allocf(FD);
    float* ub       = allocf(FD);
    float* vloc     = allocf(FD);
    float* fh       = allocf(FD);
    float* ftmp     = allocf(FD);
    float* ft0      = allocf((size_t)FROWS * 2 * DMODEL);
    float* scb      = allocf((size_t)NB * DMODEL);
    float* tt0      = allocf((size_t)NB * DMODEL);
    float* temb     = allocf((size_t)NB * DMODEL);
    float* pred     = allocf((size_t)EROWS * 6);

    // weight convert+transpose (fp32 [K][N] -> bf16 [N][K])
    auto cvtWp = [&](const float* Wsrc, int K, int N) -> const bf16* {
        bf16* p = allocb((size_t)K * N);
        dim3 g((K + 31) / 32, (N + 31) / 32);
        k_wcvt<<<g, dim3(256), 0, stream>>>(Wsrc, p, K, N);
        return p;
    };

    auto gemm = [&](float* C, int ldC, const float* A, const bf16* Wt,
                    const float* bias, int M, int K, int N, int flags) {
        dim3 g((M + BM - 1) / BM, (N + BN - 1) / BN);
        k_gemm<<<g, dim3(128), 0, stream>>>(C, ldC, A, Wt, bias, M, K, N, flags);
    };
    auto ln = [&](float* o, const float* x, const float* res, const float* g,
                  const float* b, const float* rm, int rows, int d, int silu) {
        k_layernorm<<<rows, 256, 0, stream>>>(o, x, res, g, b, rm, d, silu);
    };
    auto seq4_face = [&](float* o, const float* x, int rows, int din, int dout,
                         int base, float* mid) {
        const bf16* w1 = cvtWp(P(base + 1), din, DMODEL);
        const bf16* w2 = cvtWp(P(base + 3), DMODEL, dout);
        gemm(mid, DMODEL, x, w1, P(base + 0), rows, din, DMODEL, 0);
        ln(mid, mid, nullptr, P(base + 5), P(base + 4), nullptr, rows, DMODEL, 1);
        gemm(o, dout, mid, w2, P(base + 2), rows, DMODEL, dout, 0);
    };

    // ---- time embedding ----
    k_sincos<<<NB, 384, 0, stream>>>(tsteps, scb);
    seq4_face(temb, scb, NB, DMODEL, DMODEL, IDX_TIME, tt0);

    // ---- surf = seq4(surfp, surfPos) + seq4(surfz, surfZ) ----
    seq4_face(surf, surfPos, FROWS, 6, DMODEL, IDX_SURFP, ft0);
    seq4_face(ftmp, surfZ, FROWS, 48, DMODEL, IDX_SURFZ, ft0);
    k_add2<<<2048, 256, 0, stream>>>(surf, surf, ftmp, FD);

    // ---- edge embed (32768 rows) ----
    {
        const bf16* w1 = cvtWp(P(IDX_EDGEP + 1), 6, DMODEL);
        const bf16* w2 = cvtWp(P(IDX_EDGEP + 3), DMODEL, DMODEL);
        gemm(t0, DMODEL, edgePos, w1, P(IDX_EDGEP + 0), EROWS, 6, DMODEL, 0);
        ln(t0, t0, nullptr, P(IDX_EDGEP + 5), P(IDX_EDGEP + 4), nullptr, EROWS, DMODEL, 1);
        gemm(t1, DMODEL, t0, w2, P(IDX_EDGEP + 2), EROWS, DMODEL, DMODEL, 0);
    }

    k_build_dense_ez<<<EROWS, 256, 0, stream>>>(dense_ez, surf, t1, temb, emask);
    k_add_bcastB<<<FROWS, 256, 0, stream>>>(h, surf, temb);
    hipMemcpyAsync(e, dense_ez, ED * sizeof(float), hipMemcpyDeviceToDevice, stream);

    // ---- layers ----
    for (int l = 0; l < NLAY; ++l) {
        const int LB = IDX_LAYERS + 32 * l;
        const bf16* Wcate = cvtWp(P(LB+1),  DMODEL, CATW);
        const bf16* Wcatf = cvtWp(P(LB+3),  DMODEL, CATW);
        const bf16* Wf1   = cvtWp(P(LB+5),  DMODEL, 2*DMODEL);
        const bf16* Wf2   = cvtWp(P(LB+7),  2*DMODEL, DMODEL);
        const bf16* Wg1   = cvtWp(P(LB+9),  DMODEL, DMODEL);
        const bf16* Wg2   = cvtWp(P(LB+11), DMODEL, DMODEL);
        const bf16* Wwe1  = cvtWp(P(LB+21), DMODEL, DMODEL);
        const bf16* Wwe2  = cvtWp(P(LB+23), DMODEL, DMODEL);
        const bf16* Wwk   = cvtWp(P(LB+25), DMODEL, DMODEL);
        const bf16* Wwo   = cvtWp(P(LB+27), DMODEL, DMODEL);
        const bf16* Wwq   = cvtWp(P(LB+29), DMODEL, DMODEL);
        const bf16* Wwv   = cvtWp(P(LB+31), DMODEL, DMODEL);
        const float *bcate = P(LB+0), *bcatf = P(LB+2), *bf1 = P(LB+4), *bf2 = P(LB+6);
        const float *bg1 = P(LB+8), *bg2 = P(LB+10);
        const float *ln0b = P(LB+12), *ln0g = P(LB+13);
        const float *ln1b = P(LB+14), *ln1g = P(LB+15);
        const float *ln2b = P(LB+16), *ln2g = P(LB+17);
        const float *lneb = P(LB+18), *lneg = P(LB+19);
        const float *bwe1 = P(LB+20), *bwe2 = P(LB+22);
        const float *bwk = P(LB+24), *bwo = P(LB+26), *bwq = P(LB+28), *bwv = P(LB+30);

        k_agg<<<FROWS, 256, 0, stream>>>(hin, h, e, emask);
        gemm(ft0, DMODEL, hin, Wg1, bg1, FROWS, DMODEL, DMODEL, FLAG_SILU);
        gemm(ftmp, DMODEL, ft0, Wg2, bg2, FROWS, DMODEL, DMODEL, 0);
        ln(h, h, ftmp, ln0g, ln0b, nullptr, FROWS, DMODEL, 0);

        gemm(qb, DMODEL, h, Wwq, bwq, FROWS, DMODEL, DMODEL, 0);
        gemm(kb, DMODEL, h, Wwk, bwk, FROWS, DMODEL, DMODEL, 0);
        gemm(vb, DMODEL, h, Wwv, bwv, FROWS, DMODEL, DMODEL, 0);
        k_attention<<<NB * NHEADS * NN, 64, 0, stream>>>(qb, kb, vb, fmask, ob);
        gemm(ftmp, DMODEL, ob, Wwo, bwo, FROWS, DMODEL, DMODEL, 0);
        ln(h, h, ftmp, ln1g, ln1b, nullptr, FROWS, DMODEL, 0);

        gemm(ft0, 2 * DMODEL, h, Wf1, bf1, FROWS, DMODEL, 2 * DMODEL, FLAG_SILU);
        gemm(ftmp, DMODEL, ft0, Wf2, bf2, FROWS, 2 * DMODEL, DMODEL, 0);
        ln(h, h, ftmp, ln2g, ln2b, fmask, FROWS, DMODEL, 0);

        gemm(ub, DMODEL, h, Wwe2, nullptr, FROWS, DMODEL, DMODEL, 0);
        gemm(t0, DMODEL, e, Wwe1, bwe1, EROWS, DMODEL, DMODEL, 0);   // heavy
        k_edge_update<<<EROWS, 256, 0, stream>>>(e, t0, ub, bwe2, emask, lneg, lneb);

        gemm(fh + (size_t)l * CATW, DMODEL, h, Wcatf, bcatf, FROWS, DMODEL, CATW, 0);
        gemm(eh + (size_t)l * CATW, DMODEL, e, Wcate, bcate, EROWS, DMODEL, CATW, 0);
    }

    // ---- fc_out2 with row-split W1 ----
    const float* W1 = P(IDX_FCOUT2 + 1);     // (2304, 768)
    const float* b1 = P(IDX_FCOUT2 + 0);
    const bf16* W1a = cvtWp(W1, DMODEL, DMODEL);
    const bf16* W1b = cvtWp(W1 + (size_t)DMODEL * DMODEL, DMODEL, DMODEL);
    const bf16* W1c = cvtWp(W1 + (size_t)2 * DMODEL * DMODEL, DMODEL, DMODEL);
    const bf16* W1l2 = cvtWp(P(IDX_FCOUT2 + 3), DMODEL, DMODEL);
    gemm(vloc, DMODEL, fh, W1c, nullptr, FROWS, DMODEL, DMODEL, 0);
    gemm(t0, DMODEL, dense_ez, W1a, nullptr, EROWS, DMODEL, DMODEL, 0);
    gemm(t0, DMODEL, eh, W1b, nullptr, EROWS, DMODEL, DMODEL, FLAG_ACCUM);
    k_fc2_combine<<<EROWS, 256, 0, stream>>>(t1, t0, vloc, b1, emask);
    ln(t1, t1, nullptr, P(IDX_FCOUT2 + 5), P(IDX_FCOUT2 + 4), nullptr, EROWS, DMODEL, 1);
    gemm(t0, DMODEL, t1, W1l2, P(IDX_FCOUT2 + 2), EROWS, DMODEL, DMODEL, 0);

    // ---- fc_pool ----
    {
        const bf16* wp1 = cvtWp(P(IDX_FCPOOL + 1), DMODEL, DMODEL);
        const bf16* wp2 = cvtWp(P(IDX_FCPOOL + 3), DMODEL, DMODEL);
        gemm(t1, DMODEL, t0, wp1, P(IDX_FCPOOL + 0), EROWS, DMODEL, DMODEL, 0);
        ln(t1, t1, nullptr, P(IDX_FCPOOL + 5), P(IDX_FCPOOL + 4), nullptr, EROWS, DMODEL, 1);
        gemm(t0, DMODEL, t1, wp2, P(IDX_FCPOOL + 2), EROWS, DMODEL, DMODEL, 0);
    }

    // ---- fc_out ----
    {
        const bf16* wo1 = cvtWp(P(IDX_FCOUT + 1), DMODEL, DMODEL);
        const bf16* wo2 = cvtWp(P(IDX_FCOUT + 3), DMODEL, 6);
        gemm(t1, DMODEL, t0, wo1, P(IDX_FCOUT + 0), EROWS, DMODEL, DMODEL, 0);
        ln(t1, t1, nullptr, P(IDX_FCOUT + 5), P(IDX_FCOUT + 4), nullptr, EROWS, DMODEL, 1);
        gemm(pred, 6, t1, wo2, P(IDX_FCOUT + 2), EROWS, DMODEL, 6, 0);
    }

    // ---- symmetrize + mask ----
    k_final<<<(EROWS * 6 + 255) / 256, 256, 0, stream>>>(out, pred, emask);
}
